// minLSTM_34479997452545
// MI455X (gfx1250) — matrix operand order repaired and checked by reading, not compiled, not run on hardware
//
#include <hip/hip_runtime.h>
#include <cmath>

// ---------------------------------------------------------------------------
// minLSTM for MI455X (gfx1250): bf16 WMMA triple-GEMM + fused gate epilogue,
// then a 3-phase parallel log-space scan (chunked cumlogsumexp).
// Round 3: async global->LDS DMA staging (ASYNCcnt path), KS=64 uniform
// staging, explicit LDS double buffering (one barrier per k-tile).
// ---------------------------------------------------------------------------

typedef __bf16 bf16;
typedef __attribute__((ext_vector_type(8)))  __bf16 v8bf;
typedef __attribute__((ext_vector_type(16))) __bf16 v16bf;
typedef __attribute__((ext_vector_type(8)))  float  v8f;

union FragU { v16bf v; v8bf h[2]; };

constexpr int BATCH = 4;
constexpr int SEQ   = 4096;
constexpr int KD    = 1024;   // IN
constexpr int HD    = 1024;   // H
constexpr int MT    = BATCH * SEQ;   // 16384 rows of x (flattened B*L)
constexpr int CH    = 64;            // scan chunks per channel
constexpr int CL    = SEQ / CH;      // 64 steps per chunk
constexpr int BHC   = BATCH * HD;    // 4096 channels

// GEMM tiling: block 128(M) x 32(N), 8 waves as 4(M) x 2(N), wave tile 32x16.
// KS=64: two WMMA k-substeps per staged tile; staging is uniform across 256
// threads (A: 4 x 16B chunks/thread, B: 3 x 16B chunks/thread).
constexpr int BM = 128, BN = 32, KS = 64;
constexpr int STR = 72;              // padded LDS row stride (bf16), 16B aligned
constexpr int NT  = KD / KS;         // 16 k-tiles
constexpr int ASZ = BM * STR;        // A-tile elems per buffer
constexpr int BSZ = 3 * BN * STR;    // B-tile elems per buffer

// --------------------- async global->LDS copy (CDNA5) ----------------------
// GLOBAL_LOAD_ASYNC_TO_LDS_B128: VDST = per-lane LDS byte address, VADDR =
// 64-bit global address, tracked with ASYNCcnt. Generic LDS pointers carry
// the LDS byte offset in their low 32 bits (aperture rule), so a plain cast
// yields the dsaddr operand.
__device__ __forceinline__ void async_copy_b128(void* lds_dst, const void* gsrc) {
    asm volatile("global_load_async_to_lds_b128 %0, %1, off"
                 :: "v"((unsigned)(unsigned long long)lds_dst),
                    "v"((unsigned long long)gsrc)
                 : "memory");
}
__device__ __forceinline__ void wait_async0() {
#if __has_builtin(__builtin_amdgcn_s_wait_asynccnt)
    __builtin_amdgcn_s_wait_asynccnt(0);
#else
    asm volatile("s_wait_asynccnt 0x0" ::: "memory");
#endif
}

// ---------------------------- device math helpers --------------------------

__device__ __forceinline__ float softplusf(float x) {
    return fmaxf(x, 0.0f) + log1pf(expf(-fabsf(x)));   // stable softplus
}
__device__ __forceinline__ float laef(float a, float b) {
    float m = fmaxf(a, b);
    float n = fminf(a, b);
    if (n == -INFINITY) return m;                      // -inf identity
    return m + log1pf(expf(n - m));
}
__device__ __forceinline__ float log_gf(float x) {
    return (x >= 0.0f) ? logf(x + 0.5f) : -softplusf(-x);
}

// ---------------------------- f32 -> bf16 convert --------------------------

__global__ void f32_to_bf16_kernel(const float* __restrict__ in,
                                   bf16* __restrict__ out, size_t n) {
    size_t stride = (size_t)gridDim.x * blockDim.x;
    for (size_t i = (size_t)blockIdx.x * blockDim.x + threadIdx.x; i < n; i += stride)
        out[i] = (bf16)in[i];
}

// ------------------- triple GEMM (f,i,h gates) + epilogue -------------------
// z_g = x @ W_g^T ; emits log_f and log_v = log_i + log g(z_h) directly.

__global__ __launch_bounds__(256)
__attribute__((amdgpu_waves_per_eu(2, 8)))
void gemm_gates_kernel(const bf16* __restrict__ xb,   // [MT][KD]
                       const bf16* __restrict__ wb,   // [3][HD][KD]
                       float* __restrict__ lf_out,    // [MT][HD]
                       float* __restrict__ lv_out) {  // [MT][HD]
    __shared__ __align__(16) bf16 sh[2][ASZ + BSZ];    // double-buffered, ~63 KB

    const int tid   = threadIdx.x;
    const int lane  = tid & 31;
    const int wave  = tid >> 5;
    const int waveM = wave & 3;       // 4 M-groups of 32 rows
    const int waveN = wave >> 2;      // 2 N-groups of 16 cols
    const int lrow  = lane & 15;
    const int half  = lane >> 4;      // 0: lanes 0-15, 1: lanes 16-31

    const int m0 = blockIdx.y * BM;
    const int n0 = blockIdx.x * BN;

    // Per-thread staging chunk coordinates (16B = 8 bf16 per chunk).
    // A tile: 128 rows x 64 k  -> 1024 chunks, 4 per thread.
    int arow[4], akc[4];
    #pragma unroll
    for (int j = 0; j < 4; ++j) {
        const int c = tid + j * 256;
        arow[j] = c >> 3;            // 8 chunks per 64-elem row
        akc[j]  = (c & 7) * 8;
    }
    // B tiles: 3 gates x 32 rows x 64 k -> 768 chunks, 3 per thread.
    int brow[3], bkc[3];             // brow in [0, 96)
    #pragma unroll
    for (int j = 0; j < 3; ++j) {
        const int c = tid + j * 256;
        brow[j] = c >> 3;
        bkc[j]  = (c & 7) * 8;
    }

    auto stage = [&](int p, int kt) {
        const int k0 = kt * KS;
        bf16* As = sh[p];
        bf16* Bs = sh[p] + ASZ;
        #pragma unroll
        for (int j = 0; j < 4; ++j)
            async_copy_b128(&As[arow[j] * STR + akc[j]],
                            xb + (size_t)(m0 + arow[j]) * KD + k0 + akc[j]);
        #pragma unroll
        for (int j = 0; j < 3; ++j) {
            const int g = brow[j] >> 5, n = brow[j] & 31;
            async_copy_b128(&Bs[brow[j] * STR + bkc[j]],
                            wb + ((size_t)g * HD + (n0 + n)) * KD + k0 + bkc[j]);
        }
    };

    v8f acc[3][2] = {};               // [gate][mi], f32 16x16 tiles (48 VGPRs)

    // Fragment K offsets within a 64-wide LDS row:
    //   A (16-bit 16x32): lanes 0-15 K{0..7,16..23}, lanes 16-31 K{8..15,24..31}
    //   B (16-bit 32x16): lanes 0-15 K{0..7,16..23}, lanes 16-31 K{8..15,24..31}
    const int koffA = half * 8;
    const int koffB = half * 8;

    stage(0, 0);
    wait_async0();
    __syncthreads();

    for (int kt = 0; kt < NT; ++kt) {
        const int p = kt & 1;
        if (kt + 1 < NT) stage(p ^ 1, kt + 1);   // DMA next tile while computing

        const bf16* As = sh[p];
        const bf16* Bs = sh[p] + ASZ;
        #pragma unroll
        for (int ks = 0; ks < 2; ++ks) {         // two 32-wide substeps
            const int kb = ks * 32;
            FragU afr[2];
            #pragma unroll
            for (int mi = 0; mi < 2; ++mi) {
                const int r = waveM * 32 + mi * 16 + lrow;
                afr[mi].h[0] = *(const v8bf*)(&As[r * STR + kb + koffA]);
                afr[mi].h[1] = *(const v8bf*)(&As[r * STR + kb + koffA + 16]);
            }
            #pragma unroll
            for (int g = 0; g < 3; ++g) {
                const int r = g * 32 + waveN * 16 + lrow;
                FragU bfr;
                bfr.h[0] = *(const v8bf*)(&Bs[r * STR + kb + koffB]);
                bfr.h[1] = *(const v8bf*)(&Bs[r * STR + kb + koffB + 16]);
                #pragma unroll
                for (int mi = 0; mi < 2; ++mi) {
                    acc[g][mi] = __builtin_amdgcn_wmma_f32_16x16x32_bf16(
                        false, afr[mi].v, false, bfr.v,
                        (short)0, acc[g][mi], false, false);
                }
            }
        }
        wait_async0();                // next tile landed in LDS
        __syncthreads();              // all waves done reading current tile
    }

    // ---- fused gate epilogue ----
    // C/D 16x16 f32 layout: element e -> row = base + (lane>=16 ? 8:0) + e,
    //                       col = base + (lane & 15)
    #pragma unroll
    for (int mi = 0; mi < 2; ++mi) {
        const int rbase = m0 + waveM * 32 + mi * 16 + half * 8;
        const int col   = n0 + waveN * 16 + lrow;
        const v8f zf = acc[0][mi];
        const v8f zi = acc[1][mi];
        const v8f zh = acc[2][mi];
        #pragma unroll
        for (int e = 0; e < 8; ++e) {
            const size_t idx = (size_t)(rbase + e) * HD + col;
            const float diff  = softplusf(-zf[e]) - softplusf(-zi[e]);
            const float log_f = -softplusf(diff);
            const float log_i = -softplusf(-diff);
            lf_out[idx] = log_f;
            lv_out[idx] = log_i + log_gf(zh[e]);
        }
    }
}

// ------------------------- 3-phase parallel scan ---------------------------
// Recurrence per channel: A += log_f[l]; S = lae(S, log_v[l] - A); h = exp(A+S)

__global__ void scan_phase1_kernel(const float* __restrict__ lf,
                                   const float* __restrict__ lv,
                                   float* __restrict__ Fsum,   // [CH][BHC]
                                   float* __restrict__ Tsum) { // [CH][BHC]
    const int g     = blockIdx.x * blockDim.x + threadIdx.x;
    const int ch    = g % BHC;
    const int chunk = g / BHC;
    const int b     = ch / HD;
    const int h     = ch % HD;
    const int l0    = chunk * CL;

    float A = 0.0f;
    float r = -INFINITY;
    for (int i = 0; i < CL; ++i) {
        const size_t idx = ((size_t)b * SEQ + (l0 + i)) * HD + h;
        A += lf[idx];
        r = laef(r, lv[idx] - A);
    }
    Fsum[(size_t)chunk * BHC + ch] = A;
    Tsum[(size_t)chunk * BHC + ch] = r;
}

__global__ void scan_phase2_kernel(const float* __restrict__ h0,   // [B][H]
                                   const float* __restrict__ Fsum,
                                   const float* __restrict__ Tsum,
                                   float* __restrict__ Astart,
                                   float* __restrict__ Sstart) {
    const int ch = blockIdx.x * blockDim.x + threadIdx.x;            // BHC threads
    float A = 0.0f;
    float S = log_gf(h0[ch]);                                        // log_values[0]
    for (int c = 0; c < CH; ++c) {
        const size_t j = (size_t)c * BHC + ch;
        Astart[j] = A;
        Sstart[j] = S;
        S = laef(S, Tsum[j] - A);
        A += Fsum[j];
    }
}

__global__ void scan_phase3_kernel(const float* __restrict__ lf,
                                   const float* __restrict__ lv,
                                   const float* __restrict__ Astart,
                                   const float* __restrict__ Sstart,
                                   float* __restrict__ out) {
    const int g     = blockIdx.x * blockDim.x + threadIdx.x;
    const int ch    = g % BHC;
    const int chunk = g / BHC;
    const int b     = ch / HD;
    const int h     = ch % HD;
    const int l0    = chunk * CL;

    const float As = Astart[(size_t)chunk * BHC + ch];
    const float Ss = Sstart[(size_t)chunk * BHC + ch];

    float a = 0.0f;
    float r = -INFINITY;
    for (int i = 0; i < CL; ++i) {
        const size_t idx = ((size_t)b * SEQ + (l0 + i)) * HD + h;
        a += lf[idx];
        r = laef(r, lv[idx] - a);
        const float Sl = laef(Ss, r - As);
        out[idx] = expf(As + a + Sl);
    }
}

// --------------------------------- launch ----------------------------------

extern "C" void kernel_launch(void* const* d_in, const int* in_sizes, int n_in,
                              void* d_out, int out_size, void* d_ws, size_t ws_size,
                              hipStream_t stream) {
    const float* x   = (const float*)d_in[0];   // (B, L, IN)
    const float* h0  = (const float*)d_in[1];   // (B, H)
    const float* W_f = (const float*)d_in[2];   // (H, IN)
    const float* W_i = (const float*)d_in[3];
    const float* W_h = (const float*)d_in[4];
    float* out = (float*)d_out;                 // (B, L, H)

    char* base = (char*)d_ws;
    size_t off = 0;
    bf16*  xb = (bf16*)(base + off);  off += (size_t)MT * KD * sizeof(bf16);      // 32 MB
    bf16*  wb = (bf16*)(base + off);  off += (size_t)3 * HD * KD * sizeof(bf16);  //  6 MB
    float* lf = (float*)(base + off); off += (size_t)MT * HD * sizeof(float);     // 64 MB
    float* lv = (float*)(base + off); off += (size_t)MT * HD * sizeof(float);     // 64 MB
    float* Fs = (float*)(base + off); off += (size_t)CH * BHC * sizeof(float);
    float* Ts = (float*)(base + off); off += (size_t)CH * BHC * sizeof(float);
    float* Av = (float*)(base + off); off += (size_t)CH * BHC * sizeof(float);
    float* Sv = (float*)(base + off); off += (size_t)CH * BHC * sizeof(float);
    (void)ws_size; (void)in_sizes; (void)n_in; (void)out_size;

    // 1) convert activations + weights to bf16 (lands in L2; 192 MB holds it all)
    f32_to_bf16_kernel<<<8192, 256, 0, stream>>>(x,   xb, (size_t)MT * KD);
    f32_to_bf16_kernel<<<1024, 256, 0, stream>>>(W_f, wb + 0 * (size_t)HD * KD, (size_t)HD * KD);
    f32_to_bf16_kernel<<<1024, 256, 0, stream>>>(W_i, wb + 1 * (size_t)HD * KD, (size_t)HD * KD);
    f32_to_bf16_kernel<<<1024, 256, 0, stream>>>(W_h, wb + 2 * (size_t)HD * KD, (size_t)HD * KD);

    // 2) fused triple GEMM (WMMA bf16 -> f32) + gate epilogue
    gemm_gates_kernel<<<dim3(HD / BN, MT / BM), 256, 0, stream>>>(xb, wb, lf, lv);

    // 3) chunked parallel scan over L
    scan_phase1_kernel<<<(CH * BHC) / 256, 256, 0, stream>>>(lf, lv, Fs, Ts);
    scan_phase2_kernel<<<BHC / 256, 256, 0, stream>>>(h0, Fs, Ts, Av, Sv);
    scan_phase3_kernel<<<(CH * BHC) / 256, 256, 0, stream>>>(lf, lv, Av, Sv, out);
}
